// TrDec_12635793785521
// MI455X (gfx1250) — matrix operand order repaired and checked by reading, not compiled, not run on hardware
//
#include <hip/hip_runtime.h>
#include <hip/hip_bf16.h>

// ---------------------------------------------------------------------------
// TrDec decoder for MI455X (gfx1250, wave32, WMMA).
// All GEMMs use v_wmma_f32_16x16x32_f16 (f16 inputs, f32 accumulation).
// Weights converted to f16 once per call; W_readout (33.8MB f16) stays in L2.
// ---------------------------------------------------------------------------

#define B_ 32
#define T_ 64
#define LX_ 48
#define D_ 512
#define DW_ 512
#define V_ 33000
#define VPAD_ 33024           // V padded to multiple of 16 for WMMA tiling
#define G4_ 2048              // 4*D
#define KW_ 2560              // DW + 3D + D  (word [x|h] concat K)
#define KR_ 3072              // DW + 4D + D  (rule [x|h] concat K)
#define SIX_ 3072             // 6*D readout input

typedef __attribute__((ext_vector_type(16))) _Float16 v16h;
typedef __attribute__((ext_vector_type(8)))  float    v8f;

__device__ __forceinline__ float sigmoidf_(float x) { return 1.0f / (1.0f + expf(-x)); }

// ------------------------- elementwise helpers -----------------------------

__global__ void k_f32_to_f16(const float* __restrict__ src, _Float16* __restrict__ dst, int n) {
  int i = blockIdx.x * blockDim.x + threadIdx.x;
  if (i < n) dst[i] = (_Float16)src[i];
}

__global__ void k_zero_f32(float* __restrict__ p, int n) {
  int i = blockIdx.x * blockDim.x + threadIdx.x;
  if (i < n) p[i] = 0.0f;
}

// [Wih | Whh] -> (rows, k1+k2) f16
__global__ void k_cat_weights(const float* __restrict__ Wih, int k1,
                              const float* __restrict__ Whh, int k2,
                              _Float16* __restrict__ out, int rows) {
  int cols = k1 + k2;
  int i = blockIdx.x * blockDim.x + threadIdx.x;
  if (i >= rows * cols) return;
  int r = i / cols, c = i % cols;
  float v = (c < k1) ? Wih[(size_t)r * k1 + c] : Whh[(size_t)r * k2 + (c - k1)];
  out[i] = (_Float16)v;
}

// Pad rows (V -> VPAD) and convert to f16
__global__ void k_pad_convert(const float* __restrict__ src, _Float16* __restrict__ dst,
                              int rows_src, int rows_dst, int cols) {
  int i = blockIdx.x * blockDim.x + threadIdx.x;
  if (i >= rows_dst * cols) return;
  int r = i / cols, c = i % cols;
  dst[i] = (r < rows_src) ? (_Float16)src[(size_t)r * cols + c] : (_Float16)0.0f;
}

// ------------------------------ WMMA GEMM ----------------------------------
// C(MxN) = A(MxK) @ W(NxK)^T + bias ; optional tanh ; optional f16 mirror.
// One wave per 16x16 tile, 8 waves per block. K multiple of 32, N mult of 16.
__global__ void k_gemm(const _Float16* __restrict__ A, int lda,
                       const _Float16* __restrict__ W, int ldw,
                       const float* __restrict__ bias,
                       float* __restrict__ C, int ldc,
                       _Float16* __restrict__ Ch, int ldch,
                       int M, int N, int K, int Nstore, int do_tanh) {
  int wave = (blockIdx.x * blockDim.x + threadIdx.x) >> 5;
  int lane = threadIdx.x & 31;
  int tiles_n = N >> 4;
  int tiles_m = (M + 15) >> 4;
  if (wave >= tiles_m * tiles_n) return;
  int tm = wave / tiles_n, tn = wave - tm * tiles_n;
  int m0 = tm << 4, n0 = tn << 4;
  int half = lane >> 4;          // 0: lanes 0-15, 1: lanes 16-31
  int mr = lane & 15;            // A row / B column within tile

  const _Float16* Arow = A + (size_t)(m0 + mr) * lda;
  const _Float16* Wrow = W + (size_t)(n0 + mr) * ldw + (half << 4);

  v8f acc = {};
  for (int k0 = 0; k0 < K; k0 += 32) {
    // A fragment: K = half*16 + j (contiguous, matches B)
    v16h a;
#pragma unroll
    for (int j = 0; j < 16; ++j) {
      int ka = (half << 4) + j;
      a[j] = Arow[k0 + ka];
    }
    // B fragment: lane = N column, K = half*16 + j (contiguous 32B vector)
    v16h b = *(const v16h*)(Wrow + k0);
    if (k0 + 32 < K)
      __builtin_prefetch(Wrow + k0 + 32, 0, 1);   // global_prefetch_b8
    acc = __builtin_amdgcn_wmma_f32_16x16x32_f16(
        false, a, false, b, (short)0, acc, false, false);
  }

  int n = lane & 15;
  int gn = n0 + n;
  float bv = bias ? bias[gn] : 0.0f;
#pragma unroll
  for (int r = 0; r < 8; ++r) {
    int m = (half << 3) + r;     // C/D layout: VGPR r -> M = r + 8*half
    float x = acc[r] + bv;
    if (do_tanh) x = tanhf(x);
    int gm = m0 + m;
    if (gn < Nstore) {
      if (C)  C[(size_t)gm * ldc + gn] = x;
      if (Ch) Ch[(size_t)gm * ldch + gn] = (_Float16)x;
    }
  }
}

// --------------------------- per-step kernels ------------------------------

// Gather y_emb / parent_state, assemble f16 LSTM inputs (concat [x|h]).
// w_in (B,2560): [y_emb | parent | wfeed(1024) | wh]
// r_in (B,3072): [y_emb | parent | rfeed(1024) | wh_t(filled later) | rh]
__global__ void k_build_step(const int* __restrict__ y_train, int t,
                             const float* __restrict__ emb,
                             const float* __restrict__ states,
                             const float* __restrict__ rh, const float* __restrict__ wh,
                             const float* __restrict__ rfeed, const float* __restrict__ wfeed,
                             _Float16* __restrict__ w_in, _Float16* __restrict__ r_in,
                             float* __restrict__ wmf) {
  int b = blockIdx.x;
  int tid = threadIdx.x;
  int tok = y_train[(b * T_ + t) * 3 + 0];
  int par = y_train[(b * T_ + t) * 3 + 1];
  if (tid == 0) wmf[b] = (float)y_train[(b * T_ + t) * 3 + 2];
  const float* ye = emb + (size_t)tok * DW_;
  const float* ps = states + ((size_t)par * B_ + b) * D_;
  for (int d = tid; d < D_; d += blockDim.x) {
    _Float16 e = (_Float16)ye[d];
    _Float16 p = (_Float16)ps[d];
    w_in[b * KW_ + d]            = e;
    w_in[b * KW_ + D_ + d]       = p;
    w_in[b * KW_ + 2048 + d]     = (_Float16)wh[b * D_ + d];
    r_in[b * KR_ + d]            = e;
    r_in[b * KR_ + D_ + d]       = p;
    r_in[b * KR_ + 2560 + d]     = (_Float16)rh[b * D_ + d];
  }
  for (int d = tid; d < 2 * D_; d += blockDim.x) {
    w_in[b * KW_ + 1024 + d] = (_Float16)wfeed[b * 2 * D_ + d];
    r_in[b * KR_ + 1024 + d] = (_Float16)rfeed[b * 2 * D_ + d];
  }
}

// Word LSTM pointwise + wmask blend; updates wh/wc, fills wh_t slots.
__global__ void k_lstm_word(const float* __restrict__ gates,
                            float* __restrict__ wh, float* __restrict__ wc,
                            const float* __restrict__ wmf,
                            _Float16* __restrict__ r_in, _Float16* __restrict__ inp) {
  int idx = blockIdx.x * blockDim.x + threadIdx.x;
  if (idx >= B_ * D_) return;
  int b = idx >> 9, d = idx & (D_ - 1);
  const float* g = gates + b * G4_;
  float gi = sigmoidf_(g[d]);
  float gf = sigmoidf_(g[D_ + d]);
  float gg = tanhf(g[2 * D_ + d]);
  float go = sigmoidf_(g[3 * D_ + d]);
  float c_old = wc[idx], h_old = wh[idx];
  float c_new = gf * c_old + gi * gg;
  float h_new = go * tanhf(c_new);
  float wm = wmf[b];
  float h_t = h_new * wm + h_old * (1.0f - wm);
  float c_t = c_new * wm + c_old * (1.0f - wm);
  wh[idx] = h_t;
  wc[idx] = c_t;
  r_in[b * KR_ + 2048 + d] = (_Float16)h_t;   // rule-LSTM input slot
  inp[b * SIX_ + 1536 + d] = (_Float16)h_t;   // readout slot (also q_word)
}

// Rule LSTM pointwise; updates rh/rc, writes states[t+1], fills rh_t slot.
__global__ void k_lstm_rule(const float* __restrict__ gates,
                            float* __restrict__ rh, float* __restrict__ rc,
                            float* __restrict__ states, int t,
                            _Float16* __restrict__ inp) {
  int idx = blockIdx.x * blockDim.x + threadIdx.x;
  if (idx >= B_ * D_) return;
  int b = idx >> 9, d = idx & (D_ - 1);
  const float* g = gates + b * G4_;
  float gi = sigmoidf_(g[d]);
  float gf = sigmoidf_(g[D_ + d]);
  float gg = tanhf(g[2 * D_ + d]);
  float go = sigmoidf_(g[3 * D_ + d]);
  float c_new = gf * rc[idx] + gi * gg;
  float h_new = go * tanhf(c_new);
  rh[idx] = h_new;
  rc[idx] = c_new;
  states[((size_t)(t + 1) * B_ + b) * D_ + d] = h_new;
  inp[b * SIX_ + d] = (_Float16)h_new;        // readout slot (also q_rule)
}

// MLP attention for one stream (rule/word): scores->softmax->context.
// One block per batch row; 8 waves split the LX rows, LDS for softmax.
__global__ void k_attn(const float* __restrict__ xk,    // (B,LX,D) x_enc @ W_k^T
                       const float* __restrict__ xenc,  // (B,LX,2D)
                       const float* __restrict__ qp,    // (B,D) q@Wt^T + bt
                       const float* __restrict__ Wa,    // (D)
                       const float* __restrict__ ba,    // (1)
                       const unsigned char* __restrict__ xmask,
                       float* __restrict__ feed,        // (B,2D)
                       _Float16* __restrict__ inp, int inp_off) {
  __shared__ float s_a[LX_];
  int b = blockIdx.x;
  int tid = threadIdx.x;
  int lane = tid & 31, wv = tid >> 5;
  for (int l = wv; l < LX_; l += 8) {
    float acc = 0.0f;
    const float* xr = xk + ((size_t)b * LX_ + l) * D_;
    const float* q = qp + b * D_;
    for (int d = lane; d < D_; d += 32)
      acc += tanhf(xr[d] + q[d]) * Wa[d];
    for (int off = 16; off > 0; off >>= 1)
      acc += __shfl_down(acc, off, 32);
    if (lane == 0) {
      float s = acc + ba[0];
      if (xmask[b * LX_ + l]) s = -1e9f;
      s_a[l] = s;
    }
  }
  __syncthreads();
  if (tid == 0) {
    float mx = s_a[0];
    for (int l = 1; l < LX_; ++l) mx = fmaxf(mx, s_a[l]);
    float sum = 0.0f;
    for (int l = 0; l < LX_; ++l) { float e = expf(s_a[l] - mx); s_a[l] = e; sum += e; }
    float inv = 1.0f / sum;
    for (int l = 0; l < LX_; ++l) s_a[l] *= inv;
  }
  __syncthreads();
  for (int c = tid; c < 2 * D_; c += blockDim.x) {
    float acc = 0.0f;
#pragma unroll 4
    for (int l = 0; l < LX_; ++l)
      acc += s_a[l] * xenc[((size_t)b * LX_ + l) * (2 * D_) + c];
    feed[b * 2 * D_ + c] = acc;
    inp[b * SIX_ + inp_off + c] = (_Float16)acc;
  }
}

// score = word*wm + rule*(1-wm) -> logits[b][t][v]
__global__ void k_blend(const float* __restrict__ s_rule, const float* __restrict__ s_word,
                        const float* __restrict__ wmf, float* __restrict__ out, int t) {
  int idx = blockIdx.x * blockDim.x + threadIdx.x;
  if (idx >= B_ * V_) return;
  int b = idx / V_, v = idx - b * V_;
  float wm = wmf[b];
  float sr = s_rule[(size_t)b * VPAD_ + v];
  float sw = s_word[(size_t)b * VPAD_ + v];
  out[((size_t)b * T_ + t) * V_ + v] = sw * wm + sr * (1.0f - wm);
}

// ------------------------------- host side ---------------------------------

static inline int cdiv_(int a, int b) { return (a + b - 1) / b; }

static void gemm(const _Float16* A, int lda, const _Float16* W, int ldw,
                 const float* bias, float* C, int ldc, _Float16* Ch, int ldch,
                 int M, int N, int K, int Nstore, int do_tanh, hipStream_t s) {
  int tiles = ((M + 15) / 16) * (N / 16);
  k_gemm<<<cdiv_(tiles, 8), 256, 0, s>>>(A, lda, W, ldw, bias, C, ldc, Ch, ldch,
                                         M, N, K, Nstore, do_tanh);
}

extern "C" void kernel_launch(void* const* d_in, const int* in_sizes, int n_in,
                              void* d_out, int out_size, void* d_ws, size_t ws_size,
                              hipStream_t stream) {
  // ---- inputs (setup_inputs order) ----
  const float* x_enc   = (const float*)d_in[0];        // (B,LX,2D)
  const float* dec_h0  = (const float*)d_in[1];        // (B,D)
  const float* dec_c0  = (const float*)d_in[2];        // (B,D)
  const unsigned char* x_mask = (const unsigned char*)d_in[3]; // (B,LX) bool
  const int*   y_train = (const int*)d_in[4];          // (B,T,3)
  const float* emb     = (const float*)d_in[5];        // (V,DW)
  const float* W_k     = (const float*)d_in[6];        // (D,2D)
  const float* rule_Wt = (const float*)d_in[7];
  const float* rule_bt = (const float*)d_in[8];
  const float* rule_Wa = (const float*)d_in[9];        // (1,D)
  const float* rule_ba = (const float*)d_in[10];       // (1)
  const float* word_Wt = (const float*)d_in[11];
  const float* word_bt = (const float*)d_in[12];
  const float* word_Wa = (const float*)d_in[13];
  const float* word_ba = (const float*)d_in[14];
  const float* W_rule_ro = (const float*)d_in[15];     // (D,6D)
  const float* W_word_ro = (const float*)d_in[16];
  const float* W_readout = (const float*)d_in[17];     // (V,D)
  const float* rule_Wih  = (const float*)d_in[18];     // (4D, DW+4D)
  const float* rule_Whh  = (const float*)d_in[19];     // (4D, D)
  const float* rule_b    = (const float*)d_in[20];
  const float* word_Wih  = (const float*)d_in[21];     // (4D, DW+3D)
  const float* word_Whh  = (const float*)d_in[22];
  const float* word_b    = (const float*)d_in[23];
  float* logits = (float*)d_out;                       // (B,T,V)

  // ---- workspace carve-up (256B aligned) ----
  char* ws = (char*)d_ws;
  size_t off = 0;
  auto alloc = [&](size_t bytes) -> char* {
    char* p = ws + off;
    off = (off + bytes + 255) & ~(size_t)255;
    return p;
  };
  _Float16* Wk_h   = (_Float16*)alloc((size_t)D_ * 2 * D_ * 2);
  _Float16* xenc_h = (_Float16*)alloc((size_t)B_ * LX_ * 2 * D_ * 2);
  _Float16* rWt_h  = (_Float16*)alloc((size_t)D_ * D_ * 2);
  _Float16* wWt_h  = (_Float16*)alloc((size_t)D_ * D_ * 2);
  _Float16* rRo_h  = (_Float16*)alloc((size_t)D_ * SIX_ * 2);
  _Float16* wRo_h  = (_Float16*)alloc((size_t)D_ * SIX_ * 2);
  _Float16* wCat_h = (_Float16*)alloc((size_t)G4_ * KW_ * 2);
  _Float16* rCat_h = (_Float16*)alloc((size_t)G4_ * KR_ * 2);
  _Float16* Wro_h  = (_Float16*)alloc((size_t)VPAD_ * D_ * 2);
  float* xk     = (float*)alloc((size_t)B_ * LX_ * D_ * 4);
  float* states = (float*)alloc((size_t)(T_ + 1) * B_ * D_ * 4);
  float* rh = (float*)alloc(B_ * D_ * 4);
  float* rc = (float*)alloc(B_ * D_ * 4);
  float* wh = (float*)alloc(B_ * D_ * 4);
  float* wc = (float*)alloc(B_ * D_ * 4);
  float* rfeed = (float*)alloc(B_ * 2 * D_ * 4);
  float* wfeed = (float*)alloc(B_ * 2 * D_ * 4);
  _Float16* w_in = (_Float16*)alloc(B_ * KW_ * 2);
  _Float16* r_in = (_Float16*)alloc(B_ * KR_ * 2);
  _Float16* inp  = (_Float16*)alloc(B_ * SIX_ * 2);
  float* gates_w = (float*)alloc(B_ * G4_ * 4);
  float* gates_r = (float*)alloc(B_ * G4_ * 4);
  float* qp_rule = (float*)alloc(B_ * D_ * 4);
  float* qp_word = (float*)alloc(B_ * D_ * 4);
  _Float16* t_rule = (_Float16*)alloc(B_ * D_ * 2);
  _Float16* t_word = (_Float16*)alloc(B_ * D_ * 2);
  float* s_rule = (float*)alloc((size_t)B_ * VPAD_ * 4);
  float* s_word = (float*)alloc((size_t)B_ * VPAD_ * 4);
  float* wmf = (float*)alloc(B_ * 4);
  (void)ws_size; (void)n_in; (void)in_sizes; (void)out_size;

  // ---- one-time (per call) weight conversion to f16 ----
  auto cvt = [&](const float* s, _Float16* d, int n) {
    k_f32_to_f16<<<cdiv_(n, 256), 256, 0, stream>>>(s, d, n);
  };
  cvt(W_k, Wk_h, D_ * 2 * D_);
  cvt(x_enc, xenc_h, B_ * LX_ * 2 * D_);
  cvt(rule_Wt, rWt_h, D_ * D_);
  cvt(word_Wt, wWt_h, D_ * D_);
  cvt(W_rule_ro, rRo_h, D_ * SIX_);
  cvt(W_word_ro, wRo_h, D_ * SIX_);
  k_cat_weights<<<cdiv_(G4_ * KW_, 256), 256, 0, stream>>>(word_Wih, DW_ + 3 * D_,
                                                           word_Whh, D_, wCat_h, G4_);
  k_cat_weights<<<cdiv_(G4_ * KR_, 256), 256, 0, stream>>>(rule_Wih, DW_ + 4 * D_,
                                                           rule_Whh, D_, rCat_h, G4_);
  k_pad_convert<<<cdiv_(VPAD_ * D_, 256), 256, 0, stream>>>(W_readout, Wro_h, V_, VPAD_, D_);

  // ---- state init ----
  k_zero_f32<<<cdiv_((T_ + 1) * B_ * D_, 256), 256, 0, stream>>>(states, (T_ + 1) * B_ * D_);
  k_zero_f32<<<cdiv_(B_ * 2 * D_, 256), 256, 0, stream>>>(rfeed, B_ * 2 * D_);
  k_zero_f32<<<cdiv_(B_ * 2 * D_, 256), 256, 0, stream>>>(wfeed, B_ * 2 * D_);
  hipMemcpyAsync(rh, dec_h0, B_ * D_ * 4, hipMemcpyDeviceToDevice, stream);
  hipMemcpyAsync(wh, dec_h0, B_ * D_ * 4, hipMemcpyDeviceToDevice, stream);
  hipMemcpyAsync(rc, dec_c0, B_ * D_ * 4, hipMemcpyDeviceToDevice, stream);
  hipMemcpyAsync(wc, dec_c0, B_ * D_ * 4, hipMemcpyDeviceToDevice, stream);

  // ---- x_enc_k = x_enc @ W_k^T  (M=B*LX, N=D, K=2D) ----
  gemm(xenc_h, 2 * D_, Wk_h, 2 * D_, nullptr, xk, D_, nullptr, 0,
       B_ * LX_, D_, 2 * D_, D_, 0, stream);

  // ---- sequential scan over T timesteps ----
  for (int t = 0; t < T_; ++t) {
    k_build_step<<<B_, 256, 0, stream>>>(y_train, t, emb, states, rh, wh, rfeed, wfeed,
                                         w_in, r_in, wmf);
    // word LSTM gates: (B,2560) @ (2048,2560)^T + word_b
    gemm(w_in, KW_, wCat_h, KW_, word_b, gates_w, G4_, nullptr, 0,
         B_, G4_, KW_, G4_, 0, stream);
    k_lstm_word<<<cdiv_(B_ * D_, 256), 256, 0, stream>>>(gates_w, wh, wc, wmf, r_in, inp);
    // rule LSTM gates: (B,3072) @ (2048,3072)^T + rule_b
    gemm(r_in, KR_, rCat_h, KR_, rule_b, gates_r, G4_, nullptr, 0,
         B_, G4_, KR_, G4_, 0, stream);
    k_lstm_rule<<<cdiv_(B_ * D_, 256), 256, 0, stream>>>(gates_r, rh, rc, states, t, inp);
    // attention query projections (A strided views into inp)
    gemm(inp + 0, SIX_, rWt_h, D_, rule_bt, qp_rule, D_, nullptr, 0,
         B_, D_, D_, D_, 0, stream);
    gemm(inp + 1536, SIX_, wWt_h, D_, word_bt, qp_word, D_, nullptr, 0,
         B_, D_, D_, D_, 0, stream);
    k_attn<<<B_, 256, 0, stream>>>(xk, x_enc, qp_rule, rule_Wa, rule_ba, x_mask,
                                   rfeed, inp, 512);
    k_attn<<<B_, 256, 0, stream>>>(xk, x_enc, qp_word, word_Wa, word_ba, x_mask,
                                   wfeed, inp, 2048);
    // readout projections with fused tanh -> f16
    gemm(inp, SIX_, rRo_h, SIX_, nullptr, nullptr, 0, t_rule, D_,
         B_, D_, SIX_, D_, 1, stream);
    gemm(inp, SIX_, wRo_h, SIX_, nullptr, nullptr, 0, t_word, D_,
         B_, D_, SIX_, D_, 1, stream);
    // vocab GEMMs: (B,512) @ (33024,512)^T
    gemm(t_rule, D_, Wro_h, D_, nullptr, s_rule, VPAD_, nullptr, 0,
         B_, VPAD_, D_, V_, 0, stream);
    gemm(t_word, D_, Wro_h, D_, nullptr, s_word, VPAD_, nullptr, 0,
         B_, VPAD_, D_, V_, 0, stream);
    k_blend<<<cdiv_(B_ * V_, 256), 256, 0, stream>>>(s_rule, s_word, wmf, logits, t);
  }
}
